// GatedDeltaRuleMultiStepModel_65017214926850
// MI455X (gfx1250) — matrix operand order repaired and checked by reading, not compiled, not run on hardware
//
#include <hip/hip_runtime.h>

#define BATCH 8
#define SEQ   1024
#define HEADS 8
#define CHUNK 64
#define NCHUNK 16     // SEQ / CHUNK
#define NSTEPS 32     // two sequential passes over the sequence
#define SCALE  0.125f // DK^-0.5

typedef __attribute__((ext_vector_type(16))) __bf16 v16bf;
typedef __attribute__((ext_vector_type(8)))  __bf16 v8bf;
typedef __attribute__((ext_vector_type(8)))  float  v8f;

#define LDF 65  // f32 LDS row stride (bank padding)
#define LDH 72  // bf16 LDS row stride: 144B rows -> all fragment loads 16B aligned

// One 16x16 f32 tile of D = A * B + acc over K=64.
// A stored [m][k] (row stride LDH); B stored [n][k] (row = output column).
// Per CDNA5 ISA 7.12.2: A lane=row%16, 8-wide K-groups by lane/16; B lane=col, K-group
// by lane/16; both become contiguous 16B/32B runs in this layout.
__device__ __forceinline__ v8f wmma_tile(const __bf16* __restrict__ A,
                                         const __bf16* __restrict__ B,
                                         int mr, int nc, int lane, v8f acc)
{
  const int abase = (mr + (lane & 15)) * LDH + ((lane >> 4) * 8);
  const int bbase = (nc + (lane & 15)) * LDH + ((lane >> 4) * 8);
#pragma unroll
  for (int kb = 0; kb < 64; kb += 32) {
    v8bf alo = *(const v8bf*)(A + abase + kb);
    v8bf ahi = *(const v8bf*)(A + abase + kb + 16);
    v8bf blo = *(const v8bf*)(B + bbase + kb);
    v8bf bhi = *(const v8bf*)(B + bbase + kb + 16);
    v16bf a  = __builtin_shufflevector(alo, ahi, 0,1,2,3,4,5,6,7,8,9,10,11,12,13,14,15);
    v16bf bb = __builtin_shufflevector(blo, bhi, 0,1,2,3,4,5,6,7,8,9,10,11,12,13,14,15);
    acc = __builtin_amdgcn_wmma_f32_16x16x32_bf16(false, a, false, bb,
                                                  (short)0, acc, false, false);
  }
  return acc;
}

__global__ __launch_bounds__(256) void gdr_chunked_kernel(
    const float* __restrict__ q, const float* __restrict__ k,
    const float* __restrict__ v, const float* __restrict__ g,
    const float* __restrict__ beta, const float* __restrict__ state0,
    float* __restrict__ out)
{
  // f32 buffers
  __shared__ float sSt[64][LDF];   // S^T [dk][dv], carried across all 32 chunks
  __shared__ float sMM[64][LDF];   // beta * strict-lower(Ka Kb^T), f32 for solve
  __shared__ float sR [64][LDF];   // V -> R -> substitution workspace -> O
  // bf16 WMMA operands; all read as contiguous-K fragments
  __shared__ __align__(16) __bf16 sQa  [64][LDH]; // [t][d]   A
  __shared__ __align__(16) __bf16 sKa  [64][LDH]; // [t][d]   A
  __shared__ __align__(16) __bf16 sKb  [64][LDH]; // [j][d]   B (for X Kb^T)
  __shared__ __align__(16) __bf16 sKbT [64][LDH]; // [d][t]   A (for Kb^T U)
  __shared__ __align__(16) __bf16 sS16 [64][LDH]; // [v][d]   B (for X S^T)
  __shared__ __align__(16) __bf16 sU16 [64][LDH]; // [v][t]   B (for X U)
  __shared__ __align__(16) __bf16 sP16 [64][LDH]; // [t][i]   A
  __shared__ __align__(16) __bf16 sMb16[64][LDH]; // [i][j]   A
  __shared__ float sG[64], sA[64], sAi[64], sB[64];

  const int bh   = blockIdx.x;
  const int b    = bh / HEADS;
  const int h    = bh % HEADS;
  const int tid  = threadIdx.x;
  const int lane = tid & 31;
  const int wave = tid >> 5;

  // Initial state, transposed: sSt[d][vv] = state0[b,h,vv,d]
  for (int i = tid; i < 64 * 64; i += 256) {
    int d = i >> 6, vv = i & 63;
    sSt[d][vv] = state0[(((size_t)(b * HEADS + h) * 64) + vv) * 64 + d];
  }

  for (int step = 0; step < NSTEPS; ++step) {
    const int pass = step >> 4;
    const int t0   = (step & (NCHUNK - 1)) * CHUNK;
    __syncthreads();

    // Gate / beta staging (single coalesced-ish global read)
    if (tid < CHUNK) {
      sG[tid] = g[(size_t)(b * SEQ + t0 + tid) * HEADS + h];
      sB[tid] = beta[(size_t)(b * SEQ + t0 + tid) * HEADS + h];
    }
    __syncthreads();
    // Inclusive cumsum of gates from LDS; exp(+/-G)
    if (tid < CHUNK) {
      float acc = 0.f;
      for (int i = 0; i <= tid; ++i) acc += sG[i];
      sA[tid]  = __expf(acc);
      sAi[tid] = __expf(-acc);
    }
    __syncthreads();

    // Build bf16 operands (coalesced q/k/v reads)
    for (int i = tid; i < CHUNK * 64; i += 256) {
      int t = i >> 6, d = i & 63;
      size_t gi = ((size_t)(b * SEQ + t0 + t) * HEADS + h) * 64 + d;
      float a = sA[t], ai = sAi[t];
      float kv = k[gi];
      __bf16 kb16 = (__bf16)(kv * ai);
      sQa[t][d]  = (__bf16)(q[gi] * (a * SCALE));
      sKa[t][d]  = (__bf16)(kv * a);
      sKb[t][d]  = kb16;
      sKbT[d][t] = kb16;
      sR[t][d]   = v[gi];              // stash V
      sS16[t][d] = (__bf16)sSt[d][t];  // S[v][d] from S^T
      sU16[t][d] = (__bf16)0.f;        // blocked substitution relies on zeros
    }
    __syncthreads();

    // Stage 1: MM = beta ∘ tril(Ka Kb^T, -1)  and  R = beta ∘ (V - Ka S^T)
#pragma unroll
    for (int half = 0; half < 2; ++half) {
      int tt = wave + half * 8;
      int mr = (tt >> 2) * 16, nc = (tt & 3) * 16;
      v8f z = {0,0,0,0,0,0,0,0};
      v8f mAcc = wmma_tile(&sKa[0][0], &sKb[0][0],  mr, nc, lane, z);
      v8f rAcc = wmma_tile(&sKa[0][0], &sS16[0][0], mr, nc, lane, z);
      int col = nc + (lane & 15);
      int rb  = mr + (lane >> 4) * 8;
#pragma unroll
      for (int e = 0; e < 8; ++e) {
        int r = rb + e;
        float bet = sB[r];
        float sel = (col < r) ? 1.0f : 0.0f;   // branchless mask
        float mv  = bet * mAcc[e] * sel;
        sMM[r][col]   = mv;
        sMb16[r][col] = (__bf16)mv;
        sR[r][col]    = bet * (sR[r][col] - rAcc[e]);
      }
    }
    __syncthreads();

    // Stage 2: blocked forward substitution U = (I + M)^{-1} R, 16-row blocks
    for (int blk = 0; blk < 4; ++blk) {
      int mr = blk * 16;
      if (blk > 0 && wave < 4) {
        // Subtract finished-block contributions (unfinished U rows are zero)
        v8f z = {0,0,0,0,0,0,0,0};
        v8f tAcc = wmma_tile(&sMb16[0][0], &sU16[0][0], mr, wave * 16, lane, z);
        int col = wave * 16 + (lane & 15);
        int rb  = mr + (lane >> 4) * 8;
#pragma unroll
        for (int e = 0; e < 8; ++e) sR[rb + e][col] -= tAcc[e];
      }
      __syncthreads();
      // 16-step diagonal solve; one lane per output column, in registers
      if (tid < 64) {
        float u[16];
#pragma unroll
        for (int ii = 0; ii < 16; ++ii) {
          float x = sR[mr + ii][tid];
#pragma unroll
          for (int j = 0; j < ii; ++j) x -= sMM[mr + ii][mr + j] * u[j];
          u[ii] = x;
        }
#pragma unroll
        for (int ii = 0; ii < 16; ++ii)
          sU16[tid][mr + ii] = (__bf16)u[ii];   // [v][t]: contiguous writeback
      }
      __syncthreads();
    }

    // Stage 3: P = tril-inclusive(Qa Kb^T) -> bf16
#pragma unroll
    for (int half = 0; half < 2; ++half) {
      int tt = wave + half * 8;
      int mr = (tt >> 2) * 16, nc = (tt & 3) * 16;
      v8f z = {0,0,0,0,0,0,0,0};
      v8f pAcc = wmma_tile(&sQa[0][0], &sKb[0][0], mr, nc, lane, z);
      int col = nc + (lane & 15);
      int rb  = mr + (lane >> 4) * 8;
#pragma unroll
      for (int e = 0; e < 8; ++e) {
        int r = rb + e;
        float sel = (col <= r) ? 1.0f : 0.0f;
        sP16[r][col] = (__bf16)(pAcc[e] * sel);
      }
    }
    __syncthreads();

    // Stage 4: O = Qa S^T + P U (stage into sR);  S^T <- A_C * (S^T + Kb^T U)
    const float aC = sA[CHUNK - 1];
#pragma unroll
    for (int half = 0; half < 2; ++half) {
      int tt = wave + half * 8;
      int mr = (tt >> 2) * 16, nc = (tt & 3) * 16;
      v8f z = {0,0,0,0,0,0,0,0};
      v8f oAcc = wmma_tile(&sQa[0][0],  &sS16[0][0], mr, nc, lane, z);
      oAcc     = wmma_tile(&sP16[0][0], &sU16[0][0], mr, nc, lane, oAcc);
      int col = nc + (lane & 15);
      int rb  = mr + (lane >> 4) * 8;
#pragma unroll
      for (int e = 0; e < 8; ++e) sR[rb + e][col] = oAcc[e];
      v8f sAcc = wmma_tile(&sKbT[0][0], &sU16[0][0], mr, nc, lane, z);
#pragma unroll
      for (int e = 0; e < 8; ++e) {
        int r = rb + e;
        sSt[r][col] = aC * (sSt[r][col] + sAcc[e]);
      }
    }
    __syncthreads();

    // Coalesced output writeback (256B segments)
    for (int i = tid; i < CHUNK * 64; i += 256) {
      int t = i >> 6, d = i & 63;
      size_t oi = ((size_t)(b * (2 * SEQ) + pass * SEQ + t0 + t) * HEADS + h) * 64 + d;
      out[oi] = sR[t][d];
    }
  }
}

extern "C" void kernel_launch(void* const* d_in, const int* in_sizes, int n_in,
                              void* d_out, int out_size, void* d_ws, size_t ws_size,
                              hipStream_t stream) {
  (void)in_sizes; (void)n_in; (void)d_ws; (void)ws_size; (void)out_size;
  const float* q      = (const float*)d_in[0];
  const float* k      = (const float*)d_in[1];
  const float* v      = (const float*)d_in[2];
  const float* g      = (const float*)d_in[3];
  const float* beta   = (const float*)d_in[4];
  const float* state0 = (const float*)d_in[5];
  float* out = (float*)d_out;

  gdr_chunked_kernel<<<BATCH * HEADS, 256, 0, stream>>>(q, k, v, g, beta, state0, out);
}
